// Qwen3MoeSparseMoeBlock_71631464562771
// MI455X (gfx1250) — compile-verified, matrix operand order checked by reading, not run on hardware
//
#include <hip/hip_runtime.h>
#include <hip/hip_bf16.h>

#define T_TOK 4096
#define HID   2048
#define INTERN 768
#define NEXP  64
#define TOPK  8

typedef __attribute__((ext_vector_type(16))) __bf16          v16bf;
typedef __attribute__((ext_vector_type(2)))  __bf16          bf16x2;
typedef __attribute__((ext_vector_type(8)))  __bf16          bf16x8;
typedef __attribute__((ext_vector_type(8)))  float           v8f;
typedef __attribute__((ext_vector_type(4)))  float           f32x4;
typedef __attribute__((ext_vector_type(8)))  float           f32x8;
typedef __attribute__((ext_vector_type(8)))  unsigned short  u16x8;
typedef __attribute__((ext_vector_type(16))) unsigned short  u16x16;

__device__ __forceinline__ unsigned short f2bf(float f) {
    __bf16 h = (__bf16)f;                       // native v_cvt on gfx1250
    return __builtin_bit_cast(unsigned short, h);
}

// Packed pair convert: build_vector(fptrunc,fptrunc) -> v_cvt_pk_bf16_f32
__device__ __forceinline__ void store_pk_bf16(unsigned short* dst,
                                              float lo, float hi) {
    bf16x2 p;
    p[0] = (__bf16)lo;
    p[1] = (__bf16)hi;
    *(bf16x2*)dst = p;                          // one dword store
}

// ---------------------------------------------------------------------------
// Kernel 0: one-shot fp32 -> bf16 conversion of x into workspace. Saves
// re-converting each token row for each of its TOP_K expert replicas and
// lets the expert kernel use async byte-copies into LDS for the A tile.
// ---------------------------------------------------------------------------
__global__ __launch_bounds__(256) void cvt_x_kernel(
        const float* __restrict__ x, unsigned short* __restrict__ xbf) {
    const size_t i = ((size_t)blockIdx.x * 256 + threadIdx.x) * 8;
    f32x8 v = *(const f32x8*)(x + i);
    bf16x8 o;
    #pragma unroll
    for (int j = 0; j < 8; ++j) o[j] = (__bf16)v[j];
    *(bf16x8*)(xbf + i) = o;                    // 16B store, 4x v_cvt_pk
}

// ---------------------------------------------------------------------------
// Kernel 1: router. One wave per token: 32 lanes x 2 experts each compute the
// 64 logits; lane 0 does top-8 + softmax and scatters into per-expert buckets.
// ---------------------------------------------------------------------------
__global__ __launch_bounds__(256) void router_kernel(
        const float* __restrict__ x, const float* __restrict__ gw,
        float* __restrict__ logits_out, int* __restrict__ counts,
        int* __restrict__ tokIds, float* __restrict__ tokWts) {
    __shared__ float lgs[8][64];
    const int wave = threadIdx.x >> 5;
    const int lane = threadIdx.x & 31;
    const int t = blockIdx.x * 8 + wave;

    const float* xr = x + (size_t)t * HID;
    float a0 = 0.f, a1 = 0.f;
    for (int h = 0; h < HID; ++h) {
        float xv = xr[h];
        a0 += xv * gw[h * NEXP + lane];
        a1 += xv * gw[h * NEXP + 32 + lane];
    }
    logits_out[t * NEXP + lane]      = a0;
    logits_out[t * NEXP + 32 + lane] = a1;
    lgs[wave][lane]      = a0;
    lgs[wave][lane + 32] = a1;
    __syncthreads();

    if (lane == 0) {
        float vals[NEXP];
        #pragma unroll
        for (int i = 0; i < NEXP; ++i) vals[i] = lgs[wave][i];
        int   idx[TOPK];
        float w[TOPK];
        for (int k = 0; k < TOPK; ++k) {
            float m = -3.4e38f; int mi = 0;
            for (int i = 0; i < NEXP; ++i)
                if (vals[i] > m) { m = vals[i]; mi = i; }
            idx[k] = mi; w[k] = m; vals[mi] = -3.4e38f;
        }
        float mx = w[0], s = 0.f;
        for (int k = 0; k < TOPK; ++k) { w[k] = __expf(w[k] - mx); s += w[k]; }
        float inv = __builtin_amdgcn_rcpf(s);
        for (int k = 0; k < TOPK; ++k) {
            int e = idx[k];
            int pos = atomicAdd(&counts[e], 1);
            tokIds[e * T_TOK + pos] = t;
            tokWts[e * T_TOK + pos] = w[k] * inv;
        }
    }
}

// ---------------------------------------------------------------------------
// Fragment helpers (wave32 WMMA layouts per CDNA5 ISA 7.12.2).
// A (16x32 bf16): lanes 0-15 row=lane, K {kb..kb+7, kb+16..kb+23};
//                 lanes 16-31 row=lane-16, K {kb+8..kb+15, kb+24..kb+31}.
// B is staged into LDS already in per-lane fragment order -> contiguous read.
// ---------------------------------------------------------------------------
__device__ __forceinline__ v16bf loadAfrag(const unsigned short* base, int lane,
                                           int kbase, int stride) {
    const int row = lane & 15;
    const int k0  = kbase + ((lane >> 4) << 3);
    u16x8 lo = *(const u16x8*)(base + row * stride + k0);
    u16x8 hi = *(const u16x8*)(base + row * stride + k0 + 16);
    u16x16 r;
    #pragma unroll
    for (int i = 0; i < 8; ++i) { r[i] = lo[i]; r[i + 8] = hi[i]; }
    return __builtin_bit_cast(v16bf, r);
}

__device__ __forceinline__ v16bf loadBfrag(const unsigned short* b, int lane) {
    u16x16 r = *(const u16x16*)(b + lane * 16);
    return __builtin_bit_cast(v16bf, r);
}

// ---- register-double-buffered B-tile staging -----------------------------
// A 32(K) x 16(N) fp32 tile is held as 4 float4 per lane (2 K-pairs x 4 N).
// loadBtile issues the global loads; storeBtile converts (two bf16 packed
// into one dword -> v_cvt_pk_bf16_f32 + ds_store_b32) into fragment order.
__device__ __forceinline__ void loadBtile(const float* __restrict__ W, int stride,
                                          int kbase, int nbase, int lane,
                                          f32x4 r[4]) {
    #pragma unroll
    for (int it = 0; it < 2; ++it) {
        const int k  = it * 16 + ((lane >> 2) << 1);   // even k
        const int nq = (lane & 3) << 2;
        const float* src = W + (size_t)(kbase + k) * stride + nbase + nq;
        __builtin_prefetch((const void*)(src + (size_t)64 * stride), 0, 1);
        r[it * 2 + 0] = *(const f32x4*)src;
        r[it * 2 + 1] = *(const f32x4*)(src + stride);
    }
}

__device__ __forceinline__ void storeBtile(const f32x4 r[4], int lane,
                                           unsigned short* dst) {
    #pragma unroll
    for (int it = 0; it < 2; ++it) {
        const int k     = it * 16 + ((lane >> 2) << 1);
        const int nq    = (lane & 3) << 2;
        const int dlgrp = ((k >> 3) & 1) << 4;
        const int di    = (k & 7) | ((k >> 4) << 3);   // even
        #pragma unroll
        for (int j = 0; j < 4; ++j) {
            const int dl = nq + j + dlgrp;
            store_pk_bf16(dst + dl * 16 + di, r[it * 2][j], r[it * 2 + 1][j]);
        }
    }
}

// ---------------------------------------------------------------------------
// Kernel 2: per-(expert, 16-token tile) fused gate/up/silu/down with WMMA.
// 8 waves/block; dynamic LDS: A tile 16x2048 bf16 (64KB, async-filled) +
// per-wave B staging (16KB) + activation tile 16x768 bf16 (24KB).
// ---------------------------------------------------------------------------
__global__ __launch_bounds__(256, 1) void moe_expert_kernel(
        const unsigned short* __restrict__ xbf,
        const float* __restrict__ gate_proj,
        const float* __restrict__ up_proj,
        const float* __restrict__ down_proj,
        const int*  __restrict__ counts,
        const int*  __restrict__ tokIds,
        const float* __restrict__ tokWts,
        float* __restrict__ out) {
    extern __shared__ __attribute__((aligned(128))) unsigned short smem[];
    unsigned short* ldsA   = smem;                        // 16*2048 bf16
    unsigned short* ldsB   = smem + 16 * HID;             // 8 * 2 * 512 bf16
    unsigned short* ldsAct = smem + 16 * HID + 8192;      // 16*768 bf16
    __shared__ int   sTok[16];
    __shared__ float sWt[16];

    const int e    = blockIdx.x;
    const int tile = blockIdx.y;
    const int cnt  = counts[e];
    if (tile * 16 >= cnt) return;

    const int tid  = threadIdx.x;
    const int wave = tid >> 5;
    const int lane = tid & 31;

    if (tid < 16) {
        int idx = tile * 16 + tid;
        int tok = -1; float w = 0.f;
        if (idx < cnt) { tok = tokIds[e * T_TOK + idx]; w = tokWts[e * T_TOK + idx]; }
        sTok[tid] = tok; sWt[tid] = w;
    }
    __syncthreads();

    // ---- Phase 1: async-copy X tile (16 x 2048 bf16) into LDS ----
    #pragma unroll 1
    for (int j = 0; j < 16; ++j) {
        int tok  = sTok[j];
        int tokc = tok < 0 ? 0 : tok;          // rows with weight 0 get junk
        unsigned lds_off = (unsigned)(unsigned long long)
                           (ldsA + j * HID + tid * 8);
        const unsigned short* g = xbf + (size_t)tokc * HID + tid * 8;
        asm volatile("global_load_async_to_lds_b128 %0, %1, off"
                     :: "v"(lds_off), "v"(g) : "memory");
    }
    asm volatile("s_wait_asynccnt 0" ::: "memory");
    __syncthreads();

    const float* gp = gate_proj + (size_t)e * HID * INTERN;
    const float* up = up_proj   + (size_t)e * HID * INTERN;
    const float* dp = down_proj + (size_t)e * INTERN * HID;
    unsigned short* bG = ldsB + wave * 1024;
    unsigned short* bU = bG + 512;

    // ---- Phase 2: gate & up GEMMs (16 x 2048) @ (2048 x 768), 6 N-tiles/wave
    #pragma unroll 1
    for (int ni = 0; ni < 6; ++ni) {
        const int nbase = (wave + ni * 8) * 16;
        v8f accG = {0.f,0.f,0.f,0.f,0.f,0.f,0.f,0.f};
        v8f accU = {0.f,0.f,0.f,0.f,0.f,0.f,0.f,0.f};
        f32x4 rg[4], ru[4];
        loadBtile(gp, INTERN, 0, nbase, lane, rg);
        loadBtile(up, INTERN, 0, nbase, lane, ru);
        #pragma unroll 1
        for (int kk = 0; kk < 64; ++kk) {
            const int kbase = kk * 32;
            storeBtile(rg, lane, bG);
            storeBtile(ru, lane, bU);
            if (kk < 63) {                      // prefetch next K-tile to regs
                loadBtile(gp, INTERN, kbase + 32, nbase, lane, rg);
                loadBtile(up, INTERN, kbase + 32, nbase, lane, ru);
            }
            __builtin_amdgcn_wave_barrier();
            v16bf a  = loadAfrag(ldsA, lane, kbase, HID);
            v16bf bg = loadBfrag(bG, lane);
            v16bf bu = loadBfrag(bU, lane);
            accG = __builtin_amdgcn_wmma_f32_16x16x32_bf16(
                       false, a, false, bg, (short)0, accG, false, false);
            accU = __builtin_amdgcn_wmma_f32_16x16x32_bf16(
                       false, a, false, bu, (short)0, accU, false, false);
            __builtin_amdgcn_wave_barrier();
        }
        // silu(gate)*up -> activation tile in LDS (own columns, no conflicts)
        const int col = nbase + (lane & 15);
        #pragma unroll
        for (int j = 0; j < 8; ++j) {
            float g   = accG[j];
            float sig = __builtin_amdgcn_rcpf(1.f + __expf(-g));
            float act = g * sig * accU[j];
            const int row = j + ((lane >> 4) << 3);
            ldsAct[row * INTERN + col] = f2bf(act);
        }
    }
    __syncthreads();   // all activation columns visible to all waves

    // ---- Phase 3: down GEMM (16 x 768) @ (768 x 2048), 16 N-tiles/wave ----
    #pragma unroll 1
    for (int ni = 0; ni < 16; ++ni) {
        const int nbase = (wave + ni * 8) * 16;
        v8f acc = {0.f,0.f,0.f,0.f,0.f,0.f,0.f,0.f};
        f32x4 rd[4];
        loadBtile(dp, HID, 0, nbase, lane, rd);
        #pragma unroll 1
        for (int kk = 0; kk < 24; ++kk) {
            const int kbase = kk * 32;
            storeBtile(rd, lane, bG);
            if (kk < 23)
                loadBtile(dp, HID, kbase + 32, nbase, lane, rd);
            __builtin_amdgcn_wave_barrier();
            v16bf a = loadAfrag(ldsAct, lane, kbase, INTERN);
            v16bf b = loadBfrag(bG, lane);
            acc = __builtin_amdgcn_wmma_f32_16x16x32_bf16(
                      false, a, false, b, (short)0, acc, false, false);
            __builtin_amdgcn_wave_barrier();
        }
        const int col = nbase + (lane & 15);
        #pragma unroll
        for (int j = 0; j < 8; ++j) {
            const int row = j + ((lane >> 4) << 3);
            const int tok = sTok[row];
            if (tok >= 0)
                atomicAdd(out + (size_t)tok * HID + col, acc[j] * sWt[row]);
        }
    }
}

// ---------------------------------------------------------------------------
extern "C" void kernel_launch(void* const* d_in, const int* in_sizes, int n_in,
                              void* d_out, int out_size, void* d_ws, size_t ws_size,
                              hipStream_t stream) {
    (void)in_sizes; (void)n_in; (void)out_size; (void)ws_size;
    const float* x         = (const float*)d_in[0];
    const float* gate_w    = (const float*)d_in[1];
    const float* gate_proj = (const float*)d_in[2];
    const float* up_proj   = (const float*)d_in[3];
    const float* down_proj = (const float*)d_in[4];

    float* out        = (float*)d_out;                    // [T, H]
    float* logits_out = out + (size_t)T_TOK * HID;        // [T, E] (tuple part 2)

    int*            counts = (int*)d_ws;                       // [64]
    int*            tokIds = counts + NEXP;                    // [64][4096]
    float*          tokWts = (float*)(tokIds + NEXP * T_TOK);  // [64][4096]
    unsigned short* xbf    = (unsigned short*)(tokWts + NEXP * T_TOK); // [T][H] bf16

    (void)hipMemsetAsync(out, 0, (size_t)T_TOK * HID * sizeof(float), stream);
    (void)hipMemsetAsync(counts, 0, NEXP * sizeof(int), stream);

    cvt_x_kernel<<<(T_TOK * HID) / (256 * 8), 256, 0, stream>>>(x, xbf);
    router_kernel<<<T_TOK / 8, 256, 0, stream>>>(x, gate_w, logits_out,
                                                 counts, tokIds, tokWts);

    const size_t SMEM = (size_t)(16 * HID + 8 * 2 * 512 + 16 * INTERN)
                        * sizeof(unsigned short);
    (void)hipFuncSetAttribute((const void*)moe_expert_kernel,
                              hipFuncAttributeMaxDynamicSharedMemorySize, (int)SMEM);
    moe_expert_kernel<<<dim3(NEXP, 256), 256, SMEM, stream>>>(
        xbf, gate_proj, up_proj, down_proj, counts, tokIds, tokWts, out);
}
